// LSTMClassifier_22428319219874
// MI455X (gfx1250) — matrix operand order repaired and checked by reading, not compiled, not run on hardware
//
#include <hip/hip_runtime.h>
#include <hip/hip_bf16.h>

typedef __bf16 bf16_t;
typedef __attribute__((ext_vector_type(16))) __bf16 v16bf;
typedef __attribute__((ext_vector_type(8)))  __bf16 v8bf;
typedef __attribute__((ext_vector_type(4)))  __bf16 v4bf;
typedef __attribute__((ext_vector_type(8)))  float  v8f;
typedef __attribute__((ext_vector_type(4)))  float  v4f;

#define TSTEPS 2048
#define HID    128
#define BATCH  256

__device__ __forceinline__ float fast_sigmoid(float x) {
    return 1.0f / (1.0f + __expf(-x));
}
__device__ __forceinline__ float fast_tanh(float x) {
    // 1 - 2/(e^{2x}+1): exact limits at +/-inf, one v_exp_f32
    return 1.0f - 2.0f / (__expf(2.0f * x) + 1.0f);
}

__device__ __forceinline__ v16bf cat8(v8bf lo, v8bf hi) {
    return __builtin_shufflevector(lo, hi, 0,1,2,3,4,5,6,7,8,9,10,11,12,13,14,15);
}

// Load one 32x16 (KxN) WMMA B-operand fragment from row-major f32 W [Ntot x Ktot].
// Layout: lanes 0-15 hold column n0+lane, K=k0..k0+7,k0+16..k0+23; lanes 16-31 hold K=k0+8..k0+15,k0+24..k0+31.
template<int SK>
__device__ __forceinline__ v16bf load_wfrag(const float* __restrict__ W, int n0, int k0,
                                            int row, int hi) {
    const float* p = W + (size_t)(n0 + row) * SK + k0 + hi * 8;
    v4f a = *(const v4f*)(p + 0);
    v4f b = *(const v4f*)(p + 4);
    v4f c = *(const v4f*)(p + 16);
    v4f d = *(const v4f*)(p + 20);
    v16bf r;
    #pragma unroll
    for (int i = 0; i < 4; ++i) {
        r[i +  0] = (bf16_t)a[i];
        r[i +  4] = (bf16_t)b[i];
        r[i +  8] = (bf16_t)c[i];
        r[i + 12] = (bf16_t)d[i];
    }
    return r;
}

// Persistent LSTM layer: one workgroup per 16-row batch tile, 8 waves.
// Wave j owns H-columns [16j,16j+16) and the 4 gate tiles (i,f,g,o) above them.
// Weights live in VGPRs; h_{t-1} is exchanged via double-buffered LDS.
template<int DIN, bool WRITE_ALL>
__global__ __launch_bounds__(256)
void lstm_layer_kernel(const bf16_t* __restrict__ xin,   // [B, T, DIN] bf16
                       const float*  __restrict__ w_ih,  // [512, DIN]
                       const float*  __restrict__ w_hh,  // [512, 128]
                       const float*  __restrict__ b_ih,  // [512]
                       const float*  __restrict__ b_hh,  // [512]
                       bf16_t*       __restrict__ hout,  // [B, T, 128] bf16 (if WRITE_ALL)
                       float*        __restrict__ hlast) // [B, 128] f32 (t = T-1)
{
    constexpr int NKX  = DIN / 32;   // k-steps for the x part
    constexpr int HROW = 136;        // padded LDS row stride (halves): 272B, bank-conflict-free b128 reads
    __shared__ __align__(16) bf16_t hbuf[2][16 * HROW];

    const int tid  = threadIdx.x;
    const int lane = tid & 31;
    const int wave = tid >> 5;          // column group 0..7
    const int row  = lane & 15;
    const int hi   = (lane >> 4) & 1;   // lane half selects K sub-block / C rows
    const int b0   = blockIdx.x * 16;

    // ---- register-resident weight fragments + bias ----
    v16bf wx[4][NKX];
    v16bf wh[4][4];
    float bias[4];
    #pragma unroll
    for (int g = 0; g < 4; ++g) {
        const int n0 = g * HID + wave * 16;
        #pragma unroll
        for (int k = 0; k < NKX; ++k) wx[g][k] = load_wfrag<DIN>(w_ih, n0, k * 32, row, hi);
        #pragma unroll
        for (int k = 0; k < 4; ++k)   wh[g][k] = load_wfrag<HID>(w_hh, n0, k * 32, row, hi);
        bias[g] = b_ih[n0 + row] + b_hh[n0 + row];
    }

    // h_{-1} = 0
    for (int i = tid; i < 16 * HROW; i += 256) hbuf[0][i] = (bf16_t)0.0f;
    __syncthreads();

    const bf16_t* xrow = xin + (size_t)(b0 + row) * TSTEPS * DIN;

    v8f cst = {0.f, 0.f, 0.f, 0.f, 0.f, 0.f, 0.f, 0.f};

    // prefetch t=0 x A-fragments (A: lane m = row m; halves K {0..7} & {16..23} (+8 for hi lanes))
    v16bf xcur[NKX];
    #pragma unroll
    for (int k = 0; k < NKX; ++k) {
        const bf16_t* p = xrow + k * 32 + hi * 8;
        xcur[k] = cat8(*(const v8bf*)p, *(const v8bf*)(p + 16));
    }

    int pb = 0;
    for (int t = 0; t < TSTEPS; ++t) {
        // prefetch next timestep's x fragments; latency hides under the WMMA chain
        const int tn = (t + 1 < TSTEPS) ? (t + 1) : t;
        v16bf xnxt[NKX];
        #pragma unroll
        for (int k = 0; k < NKX; ++k) {
            const bf16_t* p = xrow + (size_t)tn * DIN + k * 32 + hi * 8;
            xnxt[k] = cat8(*(const v8bf*)p, *(const v8bf*)(p + 16));
        }

        // accumulators start at bias (bias depends only on the lane's column)
        v8f acc[4];
        #pragma unroll
        for (int g = 0; g < 4; ++g) {
            #pragma unroll
            for (int i = 0; i < 8; ++i) acc[g][i] = bias[g];
        }

        // input contribution: x_t[16 x DIN] * W_ih^T
        #pragma unroll
        for (int k = 0; k < NKX; ++k) {
            v16bf a = xcur[k];
            #pragma unroll
            for (int g = 0; g < 4; ++g)
                acc[g] = __builtin_amdgcn_wmma_f32_16x16x32_bf16(
                    false, a, false, wx[g][k], (short)0, acc[g], false, false);
        }
        // recurrent contribution: h_{t-1}[16 x 128] * W_hh^T  (A fragments from LDS)
        const bf16_t* hp = &hbuf[pb][row * HROW];
        #pragma unroll
        for (int k = 0; k < 4; ++k) {
            const bf16_t* p = hp + k * 32 + hi * 8;
            v16bf a = cat8(*(const v8bf*)p, *(const v8bf*)(p + 16));
            #pragma unroll
            for (int g = 0; g < 4; ++g)
                acc[g] = __builtin_amdgcn_wmma_f32_16x16x32_bf16(
                    false, a, false, wh[g][k], (short)0, acc[g], false, false);
        }

        // gates + state update; lane owns column wave*16+row, rows hi*8 + i (C/D layout)
        v8f hnew;
        #pragma unroll
        for (int i = 0; i < 8; ++i) {
            float iv = fast_sigmoid(acc[0][i]);
            float fv = fast_sigmoid(acc[1][i]);
            float gv = fast_tanh(acc[2][i]);
            float ov = fast_sigmoid(acc[3][i]);
            float cv = fv * cst[i] + iv * gv;
            cst[i]  = cv;
            hnew[i] = ov * fast_tanh(cv);
        }

        // publish h_t (bf16) into the other LDS buffer (transpose C-layout -> row-major)
        bf16_t* hw = &hbuf[1 - pb][hi * 8 * HROW + wave * 16 + row];
        #pragma unroll
        for (int i = 0; i < 8; ++i) hw[i * HROW] = (bf16_t)hnew[i];
        __syncthreads();

        if constexpr (WRITE_ALL) {
            // coalesced b128 global store of h_t for the next layer
            const int r = tid >> 4, ch = tid & 15;
            v8bf v = *(const v8bf*)(&hbuf[1 - pb][r * HROW + ch * 8]);
            *(v8bf*)(hout + ((size_t)(b0 + r) * TSTEPS + t) * HID + ch * 8) = v;
        }
        if (t == TSTEPS - 1) {
            #pragma unroll
            for (int i = 0; i < 8; ++i)
                hlast[(size_t)(b0 + hi * 8 + i) * HID + wave * 16 + row] = hnew[i];
        }

        #pragma unroll
        for (int k = 0; k < NKX; ++k) xcur[k] = xnxt[k];
        pb ^= 1;
    }
}

__global__ void cvt_f32_bf16_kernel(const float* __restrict__ in, bf16_t* __restrict__ out, int n4) {
    int i = blockIdx.x * blockDim.x + threadIdx.x;
    if (i < n4) {
        v4f v = ((const v4f*)in)[i];
        v4bf o;
        #pragma unroll
        for (int j = 0; j < 4; ++j) o[j] = (bf16_t)v[j];
        ((v4bf*)out)[i] = o;
    }
}

__global__ void fc_kernel(const float* __restrict__ hlast, const float* __restrict__ w_fc,
                          const float* __restrict__ b_fc, float* __restrict__ out) {
    int b = blockIdx.x * blockDim.x + threadIdx.x;
    if (b < BATCH) {
        #pragma unroll
        for (int o = 0; o < 3; ++o) {
            float s = b_fc[o];
            #pragma unroll 8
            for (int k = 0; k < HID; ++k) s += hlast[b * HID + k] * w_fc[o * HID + k];
            out[b * 3 + o] = s;
        }
    }
}

extern "C" void kernel_launch(void* const* d_in, const int* in_sizes, int n_in,
                              void* d_out, int out_size, void* d_ws, size_t ws_size,
                              hipStream_t stream) {
    (void)in_sizes; (void)n_in; (void)out_size; (void)ws_size;
    const float* x    = (const float*)d_in[0];
    const float* wih0 = (const float*)d_in[1];
    const float* whh0 = (const float*)d_in[2];
    const float* bih0 = (const float*)d_in[3];
    const float* bhh0 = (const float*)d_in[4];
    const float* wih1 = (const float*)d_in[5];
    const float* whh1 = (const float*)d_in[6];
    const float* bih1 = (const float*)d_in[7];
    const float* bhh1 = (const float*)d_in[8];
    const float* wih2 = (const float*)d_in[9];
    const float* whh2 = (const float*)d_in[10];
    const float* bih2 = (const float*)d_in[11];
    const float* bhh2 = (const float*)d_in[12];
    const float* wfc  = (const float*)d_in[13];
    const float* bfc  = (const float*)d_in[14];

    char* ws = (char*)d_ws;
    size_t off = 0;
    auto carve = [&](size_t bytes) {
        void* p = ws + off;
        off += (bytes + 255) & ~(size_t)255;
        return p;
    };
    bf16_t* xbf   = (bf16_t*)carve((size_t)BATCH * TSTEPS * 64 * sizeof(bf16_t));
    bf16_t* h0    = (bf16_t*)carve((size_t)BATCH * TSTEPS * HID * sizeof(bf16_t));
    bf16_t* h1    = (bf16_t*)carve((size_t)BATCH * TSTEPS * HID * sizeof(bf16_t));
    float*  hlast = (float*) carve((size_t)BATCH * HID * sizeof(float));

    const int n4 = BATCH * TSTEPS * 64 / 4;
    cvt_f32_bf16_kernel<<<(n4 + 255) / 256, 256, 0, stream>>>(x, xbf, n4);

    lstm_layer_kernel<64,  true ><<<BATCH / 16, 256, 0, stream>>>(xbf, wih0, whh0, bih0, bhh0, h0, hlast);
    lstm_layer_kernel<128, true ><<<BATCH / 16, 256, 0, stream>>>(h0,  wih1, whh1, bih1, bhh1, h1, hlast);
    lstm_layer_kernel<128, false><<<BATCH / 16, 256, 0, stream>>>(h1,  wih2, whh2, bih2, bhh2, (bf16_t*)nullptr, hlast);

    fc_kernel<<<1, 256, 0, stream>>>(hlast, wfc, bfc, (float*)d_out);
}
